// AttentionHead_89653147337192
// MI455X (gfx1250) — compile-verified, matrix operand order checked by reading, not run on hardware
//
#include <hip/hip_runtime.h>
#include <hip/hip_bf16.h>

typedef __attribute__((ext_vector_type(16))) _Float16 v16h;
typedef __attribute__((ext_vector_type(8)))  _Float16 v8h;
typedef __attribute__((ext_vector_type(8)))  float    v8f;

#define B_BATCH 512
#define T_SEQ   256
#define C_EMB   384
#define H_DIM   64
#define QK_SCALE 0.125f          // 64^-0.5
#define NEG_INF  (-1.0e30f)

__device__ __forceinline__ v8f wmma32(v16h a, v16h b, v8f c) {
  // v_wmma_f32_16x16x32_f16: D = A(16x32 f16) * B(32x16 f16) + C(16x16 f32)
  return __builtin_amdgcn_wmma_f32_16x16x32_f16(
      /*neg_a=*/false, a, /*neg_b=*/false, b,
      /*c_mod=*/(short)0, c, /*reuse_a=*/false, /*reuse_b=*/false);
}

__device__ __forceinline__ v8h ld8(const _Float16* p) {   // 16B vector load
  return *(const v8h*)p;
}
__device__ __forceinline__ v16h cat8(v8h a, v8h b) {
  return __builtin_shufflevector(a, b, 0,1,2,3,4,5,6,7,8,9,10,11,12,13,14,15);
}

__device__ __forceinline__ float redmax16(float v) {
  #pragma unroll
  for (int m = 1; m < 16; m <<= 1) v = fmaxf(v, __shfl_xor(v, m, 16));
  return v;
}
__device__ __forceinline__ float redsum16(float v) {
  #pragma unroll
  for (int m = 1; m < 16; m <<= 1) v += __shfl_xor(v, m, 16);
  return v;
}

// ---------------------------------------------------------------------------
// Kernel 0: weight prep.  Wt[m][h][c] = (f16) W_m[c][h]   (3 x 64 x 384).
// Runs once; makes B-fragment loads in qkv_kernel contiguous 128-bit loads.
// ---------------------------------------------------------------------------
__global__ __launch_bounds__(256)
void wcvt_kernel(const float* __restrict__ Wq, const float* __restrict__ Wk,
                 const float* __restrict__ Wv, _Float16* __restrict__ Wt) {
  const int per = H_DIM * C_EMB;                 // 24576
  const int idx = blockIdx.x * 256 + threadIdx.x;
  const int m   = idx / per;
  const int rem = idx % per;
  const int h   = rem / C_EMB;
  const int c   = rem % C_EMB;
  const float* W = (m == 0) ? Wq : (m == 1) ? Wk : Wv;
  Wt[idx] = (_Float16)W[(size_t)c * H_DIM + h];
}

// ---------------------------------------------------------------------------
// Kernel 1: fused Q/K/V projection.  One 16-row tile of x per block.
// 128 threads = 4 waves; wave w computes H-columns [16w, 16w+16).
// x staged in LDS as f16 (ds_load_b128 A-fragments shared by 4 waves);
// weights come pre-transposed/f16 so B-fragments are global_load_b128.
// Q gets bias + QK_SCALE folded in; V is stored transposed [B][H][T].
// ---------------------------------------------------------------------------
__global__ __launch_bounds__(128)
void qkv_kernel(const float* __restrict__ x,
                const _Float16* __restrict__ Wt,   // [3][H_DIM][C_EMB] f16
                const float* __restrict__ bq, const float* __restrict__ bk,
                const float* __restrict__ bv,
                _Float16* __restrict__ Qh, _Float16* __restrict__ Kh,
                _Float16* __restrict__ Vt) {
  __shared__ __align__(16) _Float16 xs[16 * C_EMB];   // 12 KB
  const int row0 = blockIdx.x * 16;                   // global row (b*T + t)
  const int tid  = threadIdx.x;

  // cooperative f32 -> f16 stage of 16 rows of x
  for (int i = tid; i < 16 * C_EMB; i += 128) {
    const int r = i / C_EMB, c = i % C_EMB;
    xs[i] = (_Float16)x[(size_t)(row0 + r) * C_EMB + c];
  }
  __syncthreads();

  const int wave = tid >> 5;                   // 0..3  -> H tile
  const int lane = tid & 31;
  const int lh   = lane & 15;                  // lane within half
  const int hi   = lane >> 4;                  // half select
  const int col  = wave * 16 + lh;             // output column in [0,64)

  const _Float16* wqRow = Wt + (size_t)col * C_EMB;                 // Wq^T row
  const _Float16* wkRow = wqRow + (size_t)H_DIM * C_EMB;            // Wk^T row
  const _Float16* wvRow = wkRow + (size_t)H_DIM * C_EMB;            // Wv^T row

  v8f accq = {}, acck = {}, accv = {};
  #pragma unroll 2
  for (int ks = 0; ks < C_EMB / 32; ++ks) {    // 12 K-steps of 32
    const int k0 = ks * 32 + hi * 8;
    // A fragment (lane=M=lh; halves: k0+{0..7}, k0+16+{0..7})
    const _Float16* xrow = &xs[lh * C_EMB + k0];
    v16h a = cat8(ld8(xrow), ld8(xrow + 16));
    // B fragments: contiguous 16B loads from transposed f16 weights
    v16h fq = cat8(ld8(wqRow + k0), ld8(wqRow + k0 + 16));
    v16h fk = cat8(ld8(wkRow + k0), ld8(wkRow + k0 + 16));
    v16h fv = cat8(ld8(wvRow + k0), ld8(wvRow + k0 + 16));
    accq = wmma32(a, fq, accq);
    acck = wmma32(a, fk, acck);
    accv = wmma32(a, fv, accv);
  }

  const float biasq = bq[col], biask = bk[col], biasv = bv[col];
  const int b  = row0 / T_SEQ;                 // 16-row tile never crosses batch
  const int t0 = row0 % T_SEQ;
  #pragma unroll
  for (int r = 0; r < 8; ++r) {
    const size_t idx = (size_t)(row0 + r + hi * 8) * H_DIM + col;
    Qh[idx] = (_Float16)((accq[r] + biasq) * QK_SCALE);  // scale folded into Q
    Kh[idx] = (_Float16)(acck[r] + biask);
    // V transposed: [B][H][T] so PV B-fragments are contiguous in s
    Vt[((size_t)b * H_DIM + col) * T_SEQ + t0 + r + hi * 8] =
        (_Float16)(accv[r] + biasv);
  }
}

// ---------------------------------------------------------------------------
// Kernel 2: causal flash attention.  grid = (B, 4), block = 128 (4 waves).
// Each wave owns one 16-row Q tile of one batch.  Online softmax; K tiles
// processed in pairs of 16 so PV uses a single K=32 WMMA per H tile.
// ---------------------------------------------------------------------------
__global__ __launch_bounds__(128)
void attn_kernel(const _Float16* __restrict__ Qh,
                 const _Float16* __restrict__ Kh,
                 const _Float16* __restrict__ Vt,   // [B][H][T]
                 float* __restrict__ out) {
  __shared__ __align__(16) _Float16 pbuf[4][16 * 32];  // per-wave P staging

  const int tid  = threadIdx.x;
  const int wave = tid >> 5;
  const int lane = tid & 31;
  const int lh   = lane & 15;
  const int hi   = lane >> 4;

  const int b    = blockIdx.x;
  const int qt   = blockIdx.y * 4 + wave;      // Q tile index in [0,16)
  const int qr0  = qt * 16;                    // first Q row of the tile
  const size_t bofs  = (size_t)b * T_SEQ * H_DIM;
  const size_t bofsV = (size_t)b * H_DIM * T_SEQ;

  _Float16* myp = pbuf[wave];

  // Q A-fragments: two 16x32 chunks covering H=64 (scale already folded in)
  const _Float16* qr = Qh + bofs + (size_t)(qr0 + lh) * H_DIM + hi * 8;
  v16h qf0 = cat8(ld8(qr),      ld8(qr + 16));
  v16h qf1 = cat8(ld8(qr + 32), ld8(qr + 48));

  v8f o0 = {}, o1 = {}, o2 = {}, o3 = {};
  float mrow[8], lrow[8];
  #pragma unroll
  for (int r = 0; r < 8; ++r) { mrow[r] = NEG_INF; lrow[r] = 0.0f; }

  #pragma unroll 1
  for (int j0 = 0; j0 <= qt; j0 += 2) {        // pairs of 16-wide K tiles
    const bool have1 = (j0 + 1) <= qt;

    // ---- scores: S = Q (16x64) . K^T, per 16-wide tile, 2 WMMAs each ----
    v8f s0 = {}, s1 = {};
    {
      const _Float16* kr = Kh + bofs + (size_t)(j0 * 16 + lh) * H_DIM + hi * 8;
      s0 = wmma32(qf0, cat8(ld8(kr),      ld8(kr + 16)), s0);
      s0 = wmma32(qf1, cat8(ld8(kr + 32), ld8(kr + 48)), s0);
    }
    if (have1) {
      const _Float16* kr =
          Kh + bofs + (size_t)((j0 + 1) * 16 + lh) * H_DIM + hi * 8;
      s1 = wmma32(qf0, cat8(ld8(kr),      ld8(kr + 16)), s1);
      s1 = wmma32(qf1, cat8(ld8(kr + 32), ld8(kr + 48)), s1);
    }

    // ---- online softmax (row stats live per-lane in C layout) ----
    #pragma unroll
    for (int r = 0; r < 8; ++r) {
      const int row = r + hi * 8;              // local row within the Q tile
      float v0 = s0[r];
      float v1 = have1 ? s1[r] : NEG_INF;
      if (j0 == qt)          { if (lh > row) v0 = NEG_INF; }  // diagonal tile
      else if (j0 + 1 == qt) { if (lh > row) v1 = NEG_INF; }
      float mt   = redmax16(fmaxf(v0, v1));
      float mnew = fmaxf(mrow[r], mt);
      float sc   = __expf(mrow[r] - mnew);
      float e0   = __expf(v0 - mnew);
      float e1   = __expf(v1 - mnew);
      lrow[r] = lrow[r] * sc + redsum16(e0 + e1);
      mrow[r] = mnew;
      o0[r] *= sc; o1[r] *= sc; o2[r] *= sc; o3[r] *= sc;
      myp[row * 32 + lh]      = (_Float16)e0;  // P in row-major 16x32
      myp[row * 32 + 16 + lh] = (_Float16)e1;
    }

    // per-wave LDS RAW: DS ops are in-order per wave; make it explicit
    asm volatile("s_wait_dscnt 0" ::: "memory");

    // ---- re-read P as a 16x32 A fragment (ds_load_b128) ----
    const _Float16* pr = myp + lh * 32 + hi * 8;
    v16h pf = cat8(ld8(pr), ld8(pr + 16));

    // ---- O += P (16x32) . V (32x64), one WMMA per 16-wide H tile ----
    const int off1 = have1 ? 16 : 0;           // stay in-bounds; P half is 0
    #pragma unroll
    for (int n = 0; n < 4; ++n) {
      const _Float16* vp =
          Vt + bofsV + (size_t)(n * 16 + lh) * T_SEQ + j0 * 16 + hi * 8;
      v16h vf = cat8(ld8(vp), ld8(vp + off1)); // contiguous in s (transposed V)
      switch (n) {
        case 0: o0 = wmma32(pf, vf, o0); break;
        case 1: o1 = wmma32(pf, vf, o1); break;
        case 2: o2 = wmma32(pf, vf, o2); break;
        case 3: o3 = wmma32(pf, vf, o3); break;
      }
    }
  }

  // ---- normalize and store f32 output [B, T, H] ----
  #pragma unroll
  for (int r = 0; r < 8; ++r) {
    const float inv = 1.0f / lrow[r];
    const size_t base = ((size_t)b * T_SEQ + qr0 + r + hi * 8) * H_DIM + lh;
    out[base +  0] = o0[r] * inv;
    out[base + 16] = o1[r] * inv;
    out[base + 32] = o2[r] * inv;
    out[base + 48] = o3[r] * inv;
  }
}

// ---------------------------------------------------------------------------
extern "C" void kernel_launch(void* const* d_in, const int* in_sizes, int n_in,
                              void* d_out, int out_size, void* d_ws, size_t ws_size,
                              hipStream_t stream) {
  const float* x  = (const float*)d_in[0];
  const float* Wq = (const float*)d_in[1];
  const float* bq = (const float*)d_in[2];
  const float* Wk = (const float*)d_in[3];
  const float* bk = (const float*)d_in[4];
  const float* Wv = (const float*)d_in[5];
  const float* bv = (const float*)d_in[6];

  const size_t nQKV = (size_t)B_BATCH * T_SEQ * H_DIM;   // 8.4M elems each
  const size_t nW   = (size_t)3 * H_DIM * C_EMB;         // 73728 elems
  _Float16* Qh = (_Float16*)d_ws;                        // ~48 MB + 144 KB ws
  _Float16* Kh = Qh + nQKV;
  _Float16* Vt = Kh + nQKV;
  _Float16* Wt = Vt + nQKV;

  wcvt_kernel<<<dim3((unsigned)(nW / 256)), 256, 0, stream>>>(Wq, Wk, Wv, Wt);
  qkv_kernel<<<dim3((B_BATCH * T_SEQ) / 16), 128, 0, stream>>>(
      x, Wt, bq, bk, bv, Qh, Kh, Vt);
  attn_kernel<<<dim3(B_BATCH, 4), 128, 0, stream>>>(
      Qh, Kh, Vt, (float*)d_out);
}
